// _Weight_17351667876275
// MI455X (gfx1250) — matrix operand order repaired and checked by reading, not compiled, not run on hardware
//
#include <hip/hip_runtime.h>

// Problem constants (from reference)
#define NROWS 100000
#define D     128
#define BATCH 500000
#define KIDX  3
#define DOUT  128
#define KDIM  (KIDX * D)          // 384 = GEMM K
#define MTILE 32                   // rows of batch per block
#define NTILES (BATCH / MTILE)     // 15625 (exact)
#define PITCH 392                  // padded K pitch in bf16 elems (breaks bank-stride conflicts)
#define WLDS_BYTES (DOUT * PITCH * 2)   // 100352 B  (bf16 weight, column-major [n][k])
#define ALDS_BYTES (MTILE * PITCH * 2)  // 25088 B   (bf16 gathered A tile [m][k])
#define PBLOCKS 64                 // partial-sum blocks for x column reduction

typedef __attribute__((ext_vector_type(16))) __bf16 v16bf;
typedef __attribute__((ext_vector_type(8)))  float  v8f;

union Frag16 { v16bf v; uint4 q[2]; };

// round-to-nearest-even fp32 -> bf16, packed pair
__device__ __forceinline__ unsigned pack_bf16(float a, float b) {
    unsigned ua = __float_as_uint(a), ub = __float_as_uint(b);
    unsigned ra = (ua + 0x7FFFu + ((ua >> 16) & 1u)) >> 16;
    unsigned rb = (ub + 0x7FFFu + ((ub >> 16) & 1u)) >> 16;
    return ra | (rb << 16);
}

// ---- Step 1a: partial column sums of x (deterministic, no atomics) ----
__global__ void colsum_partial(const float* __restrict__ x,
                               float* __restrict__ partials) {
    int col = threadIdx.x;                      // 128 threads, one column each
    float acc = 0.f;
    for (int r = blockIdx.x; r < NROWS; r += PBLOCKS)
        acc += x[(size_t)r * D + col];          // coalesced: 128 lanes -> 512B rows
    partials[blockIdx.x * D + col] = acc;
}

// ---- Step 1b: reduce partials, then pool_vec = x_pool @ W[384:512] ----
__global__ void pool_finalize(const float* __restrict__ partials,
                              const float* __restrict__ weight,
                              float* __restrict__ pool_vec) {
    __shared__ float xp[D];
    int col = threadIdx.x;                      // 128 threads
    float s = 0.f;
    for (int p = 0; p < PBLOCKS; ++p) s += partials[p * D + col];
    xp[col] = s;
    __syncthreads();
    float pv = 0.f;
    for (int d = 0; d < D; ++d)
        pv += xp[d] * weight[(size_t)(KDIM + d) * DOUT + col];
    pool_vec[col] = pv;
}

// ---- Step 2: one-time W[0:384] fp32 -> bf16, transposed to [n][k] w/ pad ----
__global__ void wconv(const float* __restrict__ weight,
                      unsigned short* __restrict__ wb) {
    int i = blockIdx.x * blockDim.x + threadIdx.x;
    if (i >= DOUT * PITCH) return;
    int n = i / PITCH, k = i - n * PITCH;
    float f = (k < KDIM) ? weight[(size_t)k * DOUT + n] : 0.f;
    unsigned u = __float_as_uint(f);
    wb[i] = (unsigned short)((u + 0x7FFFu + ((u >> 16) & 1u)) >> 16);
}

// ---- Step 3: persistent gather + bf16 WMMA GEMM ----
// block = 256 threads = 8 wave32; each wave owns a 16-column slice and
// computes two 16x16 C tiles (rows 0-15 and 16-31 of the 32-row M tile).
__global__ void __launch_bounds__(256)
fused_gather_gemm(const float* __restrict__ x,
                  const long long* __restrict__ ind,
                  const unsigned short* __restrict__ wb,
                  const float* __restrict__ pool_vec,
                  float* __restrict__ out) {
    extern __shared__ char smem[];
    unsigned short* Wlds = (unsigned short*)smem;                 // [DOUT][PITCH]
    unsigned short* Albs = (unsigned short*)(smem + WLDS_BYTES);  // [MTILE][PITCH]
    const int tid = threadIdx.x;

    // Stage bf16 weights into LDS once per (persistent) block: 100 KB, b128 copies.
    {
        const uint4* src = (const uint4*)wb;
        uint4*       dst = (uint4*)Wlds;
        for (int i = tid; i < WLDS_BYTES / 16; i += 256) dst[i] = src[i];
    }

    const int lane   = tid & 31;
    const int wv     = tid >> 5;            // wave id 0..7 -> N slice
    const int col    = wv * 16 + (lane & 15);
    const int khalfA = (lane >> 4) * 8;     // A-fragment K sub-offset (ISA 16b A layout)
    const int khalfB = (lane >> 4) * 8;     // B-fragment K sub-offset (matches A layout)
    const float pv   = pool_vec[col];       // pooled term, constant per column
    __syncthreads();

    for (int t = blockIdx.x; t < NTILES; t += gridDim.x) {
        const int Mb = t * MTILE;

        // Gather 32 rows x 3 slots of x, convert fp32->bf16 into LDS A tile.
        // 96 source rows * 32 float4 = 3072 chunks -> 12 per thread, coalesced.
        for (int i = tid; i < (MTILE * KIDX) * (D / 4); i += 256) {
            int q  = i & 31;                 // float4 index within a 128-f row
            int rs = i >> 5;                 // 0..95 = (m, slot)
            int m  = rs / KIDX, s = rs - m * KIDX;
            long long r = ind[(size_t)(Mb + m) * KIDX + s];
            float4 v = ((const float4*)(x + (size_t)r * D))[q];
            unsigned p0 = pack_bf16(v.x, v.y);
            unsigned p1 = pack_bf16(v.z, v.w);
            *((uint2*)(Albs + m * PITCH + s * D + q * 4)) = make_uint2(p0, p1);
        }
        __syncthreads();

        // Init accumulators with the pooled contribution (C layout: N = lane&15).
        v8f c0, c1;
        #pragma unroll
        for (int e = 0; e < 8; ++e) { c0[e] = pv; c1[e] = pv; }

        #pragma unroll
        for (int kc = 0; kc < KDIM / 32; ++kc) {   // 12 K-chunks of 32
            Frag16 a0, a1, b;
            const unsigned short* ap0 =
                Albs + (lane & 15) * PITCH + kc * 32 + khalfA;
            const unsigned short* ap1 = ap0 + 16 * PITCH;
            a0.q[0] = *(const uint4*)ap0;          // elems 0..7  : K = base..base+7
            a0.q[1] = *(const uint4*)(ap0 + 16);   // elems 8..15 : K = base+16..+23
            a1.q[0] = *(const uint4*)ap1;
            a1.q[1] = *(const uint4*)(ap1 + 16);
            const unsigned short* bp =
                Wlds + col * PITCH + kc * 32 + khalfB;
            b.q[0] = *(const uint4*)bp;            // elems 0..7  : K = base..base+7
            b.q[1] = *(const uint4*)(bp + 16);     // elems 8..15 : K = base+16..+23

            c0 = __builtin_amdgcn_wmma_f32_16x16x32_bf16(
                    false, a0.v, false, b.v, (short)0, c0, false, false);
            c1 = __builtin_amdgcn_wmma_f32_16x16x32_bf16(
                    false, a1.v, false, b.v, (short)0, c1, false, false);
        }

        // Store: lane holds column `col`, rows (lane>>4)*8 + r (tile1 at +16).
        float* op = out + (size_t)(Mb + (lane >> 4) * 8) * DOUT + col;
        #pragma unroll
        for (int r2 = 0; r2 < 8; ++r2) {
            op[(size_t)r2 * DOUT]        = c0[r2];
            op[(size_t)(r2 + 16) * DOUT] = c1[r2];
        }
        __syncthreads();   // protect Albs before next gather overwrites it
    }
}

extern "C" void kernel_launch(void* const* d_in, const int* in_sizes, int n_in,
                              void* d_out, int out_size, void* d_ws, size_t ws_size,
                              hipStream_t stream) {
    const float*     x      = (const float*)d_in[0];
    const long long* ind    = (const long long*)d_in[1];
    const float*     weight = (const float*)d_in[2];
    float*           out    = (float*)d_out;

    // Workspace layout (floats): [pool_vec 128][partials 64*128][wb bf16 50176]
    float*          pool_vec = (float*)d_ws;
    float*          partials = pool_vec + 128;
    unsigned short* wb       = (unsigned short*)(partials + PBLOCKS * D); // 16B aligned

    colsum_partial<<<PBLOCKS, 128, 0, stream>>>(x, partials);
    pool_finalize <<<1, 128, 0, stream>>>(partials, weight, pool_vec);
    wconv<<<(DOUT * PITCH + 255) / 256, 256, 0, stream>>>(weight, wb);
    fused_gather_gemm<<<1024, 256, WLDS_BYTES + ALDS_BYTES, stream>>>(
        x, ind, wb, pool_vec, out);
}
